// KNNAttentionNoNewLayer_28458453303526
// MI455X (gfx1250) — compile-verified, matrix operand order checked by reading, not run on hardware
//
#include <hip/hip_runtime.h>
#include <hip/hip_bf16.h>

// ---------------- types ----------------
typedef __bf16 bf16_t;
typedef __attribute__((ext_vector_type(16))) __bf16 v16bf;
typedef __attribute__((ext_vector_type(8)))  float  v8f;
typedef __attribute__((ext_vector_type(4)))  unsigned int v4ui;
typedef __attribute__((ext_vector_type(8)))  int v8i;
typedef __attribute__((ext_vector_type(4)))  int v4i;

#define DMODEL 768
#define NTOK   4096      // B*S
#define NHEAD  12
#define HDIM   64
#define KNN    16
#define MDB    32768
#define DFF    3072
#define SC_WAVES 8

// ---------------- fragment loaders ----------------
// 16-bit A/B fragment layout (16x32 / 32x16): lane<16 -> row (lane&15),
// K = {k0..k0+7, k0+16..k0+23}; lane>=16 -> same row, K shifted by +8.
// Both A (row-major) and Bt (row-major, pre-transposed weights / key rows)
// load with the identical pattern: base + (lane>=16?8:0) elems, 2 x 16B.
__device__ inline v16bf load_frag_b128(const bf16_t* p) {
  union { v16bf v; float4 q[2]; } u;
  u.q[0] = *reinterpret_cast<const float4*>(p);
  u.q[1] = *reinterpret_cast<const float4*>(p + 16);
  return u.v;
}
// 8-byte aligned variant (for the TDM-padded LDS q tile, 72B row stride)
__device__ inline v16bf load_frag_b64(const bf16_t* p) {
  union { v16bf v; float2 d[4]; } u;
  u.d[0] = *reinterpret_cast<const float2*>(p);
  u.d[1] = *reinterpret_cast<const float2*>(p + 4);
  u.d[2] = *reinterpret_cast<const float2*>(p + 16);
  u.d[3] = *reinterpret_cast<const float2*>(p + 20);
  return u.v;
}
__device__ inline v8f wmma_bf16(v16bf a, v16bf b, v8f c) {
  return __builtin_amdgcn_wmma_f32_16x16x32_bf16(false, a, false, b, (short)0, c,
                                                 false, false);
}

// ---------------- weight prep ----------------
__global__ __launch_bounds__(256)
void convert_bf16_kernel(const float* __restrict__ src, bf16_t* __restrict__ dst,
                         long n) {
  long i = (long)blockIdx.x * 256 + threadIdx.x;
  if (i < n) dst[i] = (bf16_t)src[i];
}

// dst[c*R + r] = src[r*ld + coff + c]   (transpose + fp32->bf16)
__global__ __launch_bounds__(256)
void transpose_bf16_kernel(const float* __restrict__ src, bf16_t* __restrict__ dst,
                           int R, int C, int ld, int coff) {
  long i = (long)blockIdx.x * 256 + threadIdx.x;
  if (i >= (long)R * C) return;
  int r = (int)(i % R);
  int c = (int)(i / R);
  dst[(size_t)c * R + r] = (bf16_t)src[(size_t)r * ld + coff + c];
}

// ---------------- layernorm (row = 768) -> bf16 ----------------
__global__ __launch_bounds__(256)
void layernorm_bf16_kernel(const float* __restrict__ x, const float* __restrict__ g,
                           const float* __restrict__ b, bf16_t* __restrict__ out) {
  __shared__ float rs[256], rs2[256];
  const int row = blockIdx.x, tid = threadIdx.x;
  const float* xr = x + (size_t)row * DMODEL;
  float v0 = xr[tid], v1 = xr[tid + 256], v2 = xr[tid + 512];
  rs[tid]  = v0 + v1 + v2;
  rs2[tid] = v0 * v0 + v1 * v1 + v2 * v2;
  __syncthreads();
  for (int o = 128; o > 0; o >>= 1) {
    if (tid < o) { rs[tid] += rs[tid + o]; rs2[tid] += rs2[tid + o]; }
    __syncthreads();
  }
  const float mu  = rs[0] * (1.0f / DMODEL);
  const float var = rs2[0] * (1.0f / DMODEL) - mu * mu;
  const float inv = rsqrtf(var + 1e-5f);
  bf16_t* orow = out + (size_t)row * DMODEL;
  orow[tid]       = (bf16_t)((v0 - mu) * inv * g[tid]       + b[tid]);
  orow[tid + 256] = (bf16_t)((v1 - mu) * inv * g[tid + 256] + b[tid + 256]);
  orow[tid + 512] = (bf16_t)((v2 - mu) * inv * g[tid + 512] + b[tid + 512]);
}

// ---------------- generic WMMA GEMM: C[M,N] = A[M,K] * Bt[N,K]^T ----------------
// EPI 0: +bias, store f32 AND bf16 (q projection)
// EPI 1: +bias +resid, store f32   (c_proj / final proj)
// EPI 2: +bias, tanh-GELU, store bf16 (MLP fc)
template <int EPI>
__global__ __launch_bounds__(256)
void gemm_bf16_wmma(const bf16_t* __restrict__ A, const bf16_t* __restrict__ Bt,
                    const float* __restrict__ bias, const float* __restrict__ resid,
                    float* __restrict__ Cf, bf16_t* __restrict__ Cb,
                    int M, int N, int Kd) {
  __shared__ bf16_t As[128][40];   // 32-col K chunk, +8 pad (80B row stride)
  __shared__ bf16_t Bs[128][40];
  const int tid  = threadIdx.x;
  const int lane = tid & 31;
  const int w    = tid >> 5;
  const int waveM = w & 1;   // 2 x 64 rows
  const int waveN = w >> 1;  // 4 x 32 cols
  const int m0 = blockIdx.x * 128;
  const int n0 = blockIdx.y * 128;
  const int lr = lane & 15, half = lane >> 4;

  v8f acc[4][2];
#pragma unroll
  for (int mi = 0; mi < 4; ++mi)
#pragma unroll
    for (int ni = 0; ni < 2; ++ni) acc[mi][ni] = (v8f){};

  for (int k0 = 0; k0 < Kd; k0 += 32) {
#pragma unroll
    for (int i = 0; i < 2; ++i) {
      int q = tid + i * 256;           // 512 x 16B covers both 128x32 tiles
      int r = q >> 2, c8 = (q & 3) * 8;
      *reinterpret_cast<float4*>(&As[r][c8]) =
          *reinterpret_cast<const float4*>(&A[(size_t)(m0 + r) * Kd + k0 + c8]);
      *reinterpret_cast<float4*>(&Bs[r][c8]) =
          *reinterpret_cast<const float4*>(&Bt[(size_t)(n0 + r) * Kd + k0 + c8]);
    }
    __syncthreads();
    v16bf af[4], bfg[2];
#pragma unroll
    for (int mi = 0; mi < 4; ++mi)
      af[mi] = load_frag_b128(&As[waveM * 64 + mi * 16 + lr][half * 8]);
#pragma unroll
    for (int ni = 0; ni < 2; ++ni)
      bfg[ni] = load_frag_b128(&Bs[waveN * 32 + ni * 16 + lr][half * 8]);
#pragma unroll
    for (int mi = 0; mi < 4; ++mi)
#pragma unroll
      for (int ni = 0; ni < 2; ++ni)
        acc[mi][ni] = wmma_bf16(af[mi], bfg[ni], acc[mi][ni]);
    __syncthreads();
  }

#pragma unroll
  for (int mi = 0; mi < 4; ++mi)
#pragma unroll
    for (int ni = 0; ni < 2; ++ni) {
      const int col   = n0 + waveN * 32 + ni * 16 + lr;
      const int rbase = m0 + waveM * 64 + mi * 16 + half * 8;
      const float bv  = bias[col];
#pragma unroll
      for (int i = 0; i < 8; ++i) {
        const int row = rbase + i;
        float v = acc[mi][ni][i] + bv;
        const size_t off = (size_t)row * N + col;
        if (EPI == 1) v += resid[off];
        if (EPI == 2) {
          const float t = tanhf(0.7978845608028654f * (v + 0.044715f * v * v * v));
          Cb[off] = (bf16_t)(0.5f * v * (1.0f + t));
        } else {
          Cf[off] = v;
          if (EPI == 0) Cb[off] = (bf16_t)v;
        }
      }
    }
}

// ---------------- fused KNN scores + top-K ----------------
// One block = 32 query rows. TDM stages the q tile into LDS chunked as
// [24 kchunks][32 rows][32 elems + 8B pad] (72B rows -> conflict-friendly).
// 8 waves stream 16-key tiles of the bf16 key bank (48MB, L2-resident),
// 48 WMMAs per tile, maintain per-lane register top-16, then merge.
struct ScoresSmem {
  bf16_t qtile[24 * 32 * 36];                 // 55296 B
  union {
    float stage[SC_WAVES][32][16];
    struct { float s[32][SC_WAVES * KNN]; int idx[32][SC_WAVES * KNN]; } cand;
  } u;                                         // 32768 B
};

__global__ __launch_bounds__(256)
void knn_scores_topk_kernel(const bf16_t* __restrict__ qbf,
                            const bf16_t* __restrict__ dbk,
                            int* __restrict__ topk) {
  __shared__ ScoresSmem sm;
  const int tid  = threadIdx.x;
  const int lane = tid & 31;
  const int w    = tid >> 5;
  const int r0   = blockIdx.x * 32;
  const int lr = lane & 15, half = lane >> 4;

  if (w == 0) {
    // Tensor Data Mover: 3D tile (X=32 elems, Y=32 rows, Z=24 chunks),
    // pad 8B after every 64B -> LDS row stride 72B.
    const unsigned long long ga =
        (unsigned long long)(const void*)(qbf + (size_t)r0 * DMODEL);
    const unsigned int ldsOff =
        (unsigned int)(unsigned long long)(const void*)&sm.qtile[0];
    v4ui g0;
    g0[0] = 1u;                                   // count=1, user mode
    g0[1] = ldsOff;                               // lds_addr
    g0[2] = (unsigned)(ga & 0xFFFFFFFFull);       // global_addr lo
    g0[3] = (unsigned)((ga >> 32) & 0x1FFFFFFull) | (2u << 30);  // hi | type=2
    v8i g1;
    g1[0] = (1 << 16) | (1 << 20) | (3 << 22) | (1 << 25);
    //        ^2B elem   ^pad_en    ^every 64B   ^pad 8B
    g1[1] = (DMODEL & 0xFFFF) << 16;              // tensor_dim0 lo16
    g1[2] = ((DMODEL >> 16) & 0xFFFF) | ((NTOK & 0xFFFF) << 16);
    g1[3] = ((NTOK >> 16) & 0xFFFF) | (32 << 16); // tile_dim0 = 32
    g1[4] = 32 | (24 << 16);                      // tile_dim1=32, tile_dim2=24
    g1[5] = DMODEL;                               // dim0 stride
    g1[6] = (32 << 16);                           // dim1 stride = 32
    g1[7] = 0;
    v4i g2; g2[0] = 24; g2[1] = 0; g2[2] = 0; g2[3] = 0;  // tensor_dim2
    v4i g3 = {0, 0, 0, 0};
    v8i g4 = {0, 0, 0, 0, 0, 0, 0, 0};            // unused trailing group
    __builtin_amdgcn_tensor_load_to_lds(g0, g1, g2, g3, g4, 0);
    __builtin_amdgcn_s_wait_tensorcnt(0);
  }
  __syncthreads();

  float ts[KNN]; int ti[KNN];
#pragma unroll
  for (int i = 0; i < KNN; ++i) { ts[i] = -3.0e38f; ti[i] = 0; }
  float tmin = -3.0e38f; int tminpos = 0;

  for (int nt = w; nt < MDB / 16; nt += SC_WAVES) {
    const int n0 = nt * 16;
    v8f acc0 = (v8f){}, acc1 = (v8f){};
    const bf16_t* brow = dbk + (size_t)(n0 + lr) * DMODEL + half * 8;
#pragma unroll 4
    for (int c = 0; c < 24; ++c) {
      const v16bf bfrag = load_frag_b128(brow + c * 32);
      const bf16_t* ap = &sm.qtile[(size_t)(c * 32) * 36 + half * 8];
      const v16bf a0 = load_frag_b64(ap + lr * 36);
      const v16bf a1 = load_frag_b64(ap + (16 + lr) * 36);
      acc0 = wmma_bf16(a0, bfrag, acc0);
      acc1 = wmma_bf16(a1, bfrag, acc1);
    }
    // stage tile scores (wave-private; same-wave DS ordering suffices)
#pragma unroll
    for (int i = 0; i < 8; ++i) {
      sm.u.stage[w][half * 8 + i][lr]      = acc0[i];
      sm.u.stage[w][16 + half * 8 + i][lr] = acc1[i];
    }
    const float* srow = sm.u.stage[w][lane];   // lane owns query row `lane`
#pragma unroll
    for (int j = 0; j < 16; ++j) {
      const float s = srow[j];
      if (s > tmin) {
#pragma unroll
        for (int i = 0; i < KNN; ++i)
          if (i == tminpos) { ts[i] = s; ti[i] = n0 + j; }
        tmin = ts[0]; tminpos = 0;
#pragma unroll
        for (int i = 1; i < KNN; ++i)
          if (ts[i] < tmin) { tmin = ts[i]; tminpos = i; }
      }
    }
  }

  __syncthreads();   // stage region is reused as candidate buffer
#pragma unroll
  for (int i = 0; i < KNN; ++i) {
    sm.u.cand.s[lane][w * KNN + i]   = ts[i];
    sm.u.cand.idx[lane][w * KNN + i] = ti[i];
  }
  __syncthreads();

  if (tid < 32) {
    float fs[KNN]; int fi[KNN];
#pragma unroll
    for (int i = 0; i < KNN; ++i) { fs[i] = -3.0e38f; fi[i] = 0; }
    float mn = -3.0e38f; int mp = 0;
    for (int jj = 0; jj < SC_WAVES * KNN; ++jj) {
      const float s = sm.u.cand.s[tid][jj];
      const int  id = sm.u.cand.idx[tid][jj];
      if (s > mn) {
#pragma unroll
        for (int i = 0; i < KNN; ++i)
          if (i == mp) { fs[i] = s; fi[i] = id; }
        mn = fs[0]; mp = 0;
#pragma unroll
        for (int i = 1; i < KNN; ++i)
          if (fs[i] < mn) { mn = fs[i]; mp = i; }
      }
    }
#pragma unroll
    for (int i = 0; i < KNN; ++i)
      topk[(size_t)(r0 + tid) * KNN + i] = fi[i];
  }
}

// ---------------- per-token 16-way memory attention (fp32) ----------------
__global__ __launch_bounds__(256)
void knn_attention_kernel(const float* __restrict__ qf,
                          const float* __restrict__ dbkf,
                          const float* __restrict__ dbvf,
                          const int* __restrict__ topk,
                          bf16_t* __restrict__ attnb) {
  __shared__ int   idx[KNN];
  __shared__ float sc[KNN][NHEAD];
  __shared__ float aw[KNN][NHEAD];
  const int t = blockIdx.x, tid = threadIdx.x;
  if (tid < KNN) idx[tid] = topk[(size_t)t * KNN + tid];
  __syncthreads();
  if (tid < KNN * NHEAD) {
    const int m = tid / NHEAD, h = tid % NHEAD;
    const float* qp = qf + (size_t)t * DMODEL + h * HDIM;
    const float* kp = dbkf + (size_t)idx[m] * DMODEL + h * HDIM;
    float s = 0.f;
#pragma unroll 8
    for (int d = 0; d < HDIM; ++d) s += qp[d] * kp[d];
    sc[m][h] = s * 0.125f;   // 1/sqrt(64)
  }
  __syncthreads();
  if (tid < NHEAD) {
    const int h = tid;
    float mx = sc[0][h];
#pragma unroll
    for (int m = 1; m < KNN; ++m) mx = fmaxf(mx, sc[m][h]);
    float sum = 0.f;
#pragma unroll
    for (int m = 0; m < KNN; ++m) { const float e = expf(sc[m][h] - mx); aw[m][h] = e; sum += e; }
    const float inv = 1.0f / sum;
#pragma unroll
    for (int m = 0; m < KNN; ++m) aw[m][h] *= inv;
  }
  __syncthreads();
  for (int d = tid; d < DMODEL; d += 256) {
    const int h = d / HDIM;
    float a = 0.f;
#pragma unroll
    for (int m = 0; m < KNN; ++m)
      a += aw[m][h] * dbvf[(size_t)idx[m] * DMODEL + d];
    attnb[(size_t)t * DMODEL + d] = (bf16_t)a;
  }
}

// ---------------- host orchestration ----------------
extern "C" void kernel_launch(void* const* d_in, const int* in_sizes, int n_in,
                              void* d_out, int out_size, void* d_ws, size_t ws_size,
                              hipStream_t stream) {
  const float* prev    = (const float*)d_in[0];
  const float* dbK     = (const float*)d_in[1];
  const float* dbV     = (const float*)d_in[2];
  const float* ln1g    = (const float*)d_in[3];
  const float* ln1b    = (const float*)d_in[4];
  const float* cattnW  = (const float*)d_in[5];
  const float* cattnB  = (const float*)d_in[6];
  const float* cprojW  = (const float*)d_in[7];
  const float* cprojB  = (const float*)d_in[8];
  const float* ln2g    = (const float*)d_in[9];
  const float* ln2b    = (const float*)d_in[10];
  const float* fcW     = (const float*)d_in[11];
  const float* fcB     = (const float*)d_in[12];
  const float* projW   = (const float*)d_in[13];
  const float* projB   = (const float*)d_in[14];
  float* out = (float*)d_out;

  char* ws = (char*)d_ws;
  size_t off = 0;
  auto alloc = [&](size_t bytes) -> void* {
    void* p = ws + off;
    off += (bytes + 255) & ~(size_t)255;
    return p;
  };
  bf16_t* dbkB  = (bf16_t*)alloc((size_t)MDB * DMODEL * 2);
  bf16_t* wqT   = (bf16_t*)alloc((size_t)DMODEL * DMODEL * 2);
  bf16_t* cpT   = (bf16_t*)alloc((size_t)DMODEL * DMODEL * 2);
  bf16_t* fcT   = (bf16_t*)alloc((size_t)DFF * DMODEL * 2);
  bf16_t* pT    = (bf16_t*)alloc((size_t)DMODEL * DFF * 2);
  bf16_t* h1b   = (bf16_t*)alloc((size_t)NTOK * DMODEL * 2);
  float*  qf    = (float*)alloc((size_t)NTOK * DMODEL * 4);
  bf16_t* qb    = (bf16_t*)alloc((size_t)NTOK * DMODEL * 2);
  int*    tk    = (int*)alloc((size_t)NTOK * KNN * 4);
  bf16_t* attnb = (bf16_t*)alloc((size_t)NTOK * DMODEL * 2);
  float*  hmid  = (float*)alloc((size_t)NTOK * DMODEL * 4);
  bf16_t* h2b   = (bf16_t*)alloc((size_t)NTOK * DMODEL * 2);
  bf16_t* ffb   = (bf16_t*)alloc((size_t)NTOK * DFF * 2);

  // weight prep (bf16 key bank is 48MB -> L2-resident for the scores GEMM)
  {
    long n = (long)MDB * DMODEL;
    convert_bf16_kernel<<<(int)((n + 255) / 256), 256, 0, stream>>>(dbK, dbkB, n);
  }
  transpose_bf16_kernel<<<(DMODEL * DMODEL + 255) / 256, 256, 0, stream>>>(
      cattnW, wqT, DMODEL, DMODEL, 3 * DMODEL, 0);           // q slice of c_attn
  transpose_bf16_kernel<<<(DMODEL * DMODEL + 255) / 256, 256, 0, stream>>>(
      cprojW, cpT, DMODEL, DMODEL, DMODEL, 0);
  transpose_bf16_kernel<<<(DMODEL * DFF + 255) / 256, 256, 0, stream>>>(
      fcW, fcT, DMODEL, DFF, DFF, 0);
  transpose_bf16_kernel<<<(DFF * DMODEL + 255) / 256, 256, 0, stream>>>(
      projW, pT, DFF, DMODEL, DMODEL, 0);

  // LN1 -> q projection (only q of qkv is live in the reference)
  layernorm_bf16_kernel<<<NTOK, 256, 0, stream>>>(prev, ln1g, ln1b, h1b);
  gemm_bf16_wmma<0><<<dim3(NTOK / 128, DMODEL / 128), 256, 0, stream>>>(
      h1b, wqT, cattnB, nullptr, qf, qb, NTOK, DMODEL, DMODEL);

  // fused scores + top-16 over the 32768-entry memory bank
  knn_scores_topk_kernel<<<NTOK / 32, 256, 0, stream>>>(qb, dbkB, tk);

  // gather + per-head softmax attention over K=16 memories
  knn_attention_kernel<<<NTOK, 256, 0, stream>>>(qf, dbK, dbV, tk, attnb);

  // c_proj + residual
  gemm_bf16_wmma<1><<<dim3(NTOK / 128, DMODEL / 128), 256, 0, stream>>>(
      attnb, cpT, cprojB, prev, hmid, nullptr, NTOK, DMODEL, DMODEL);

  // LN2 -> MLP fc (+GELU) -> proj + residual -> d_out
  layernorm_bf16_kernel<<<NTOK, 256, 0, stream>>>(hmid, ln2g, ln2b, h2b);
  gemm_bf16_wmma<2><<<dim3(NTOK / 128, DFF / 128), 256, 0, stream>>>(
      h2b, fcT, fcB, nullptr, nullptr, ffb, NTOK, DFF, DMODEL);
  gemm_bf16_wmma<1><<<dim3(NTOK / 128, DMODEL / 128), 256, 0, stream>>>(
      ffb, pT, projB, hmid, out, nullptr, NTOK, DMODEL, DFF);
}
